// Codebook_60773787238614
// MI455X (gfx1250) — matrix operand order repaired and checked by reading, not compiled, not run on hardware
//
#include <hip/hip_runtime.h>

// CDNA5 / gfx1250 vector-quantization codebook kernel.
// dist(n,c) = ||x_n||^2 - 2 x_n.c + ||c||^2 ; argmin only needs  ||c||^2 - 2 x.c^T
// GEMM via v_wmma_f32_16x16x32_bf16 with split-bf16 (hi+lo) for ~fp32 accuracy:
//   x.c ~= xh.ch + xh.cl + xl.ch   (3 bf16 WMMA passes, f32 accumulate)
// B-chunk staging uses GLOBAL_LOAD_ASYNC_TO_LDS_B128 (ASYNCcnt) with a double
// buffer so the global->LDS copy of chunk cb+1 overlaps the WMMAs of chunk cb.

typedef __attribute__((ext_vector_type(16))) __bf16 v16bf;
typedef __attribute__((ext_vector_type(8)))  float  v8f;

#define DIMK   256   // feature dim (K)
#define NCODES 1024  // number of codes
#define CHUNK  16    // codes per LDS stage / WMMA N-tile
#define ROWQ   33    // uint4 per LDS code row: 512B data + 16B pad = 528B (bank-conflict pad)
#define BUFQ   (2 * CHUNK * ROWQ)   // uint4 per buffer (hi rows + lo rows)

// ---------------- prep: split codes into bf16 hi/lo + ||c||^2 ----------------
__global__ __launch_bounds__(256) void vq_prep(const float* __restrict__ codes,
                                               __bf16* __restrict__ ch,
                                               __bf16* __restrict__ cl,
                                               float* __restrict__ cnorm) {
  const int row = blockIdx.x;      // one code row per block
  const int t   = threadIdx.x;     // 256 threads = 256 elements
  const float v = codes[row * DIMK + t];
  const __bf16 h = (__bf16)v;
  const __bf16 l = (__bf16)(v - (float)h);
  ch[row * DIMK + t] = h;
  cl[row * DIMK + t] = l;
  float ss = v * v;
#pragma unroll
  for (int m = 16; m >= 1; m >>= 1) ss += __shfl_xor(ss, m, 32);
  __shared__ float red[8];
  if ((t & 31) == 0) red[t >> 5] = ss;
  __syncthreads();
  if (t == 0) {
    float s = 0.f;
#pragma unroll
    for (int i = 0; i < 8; ++i) s += red[i];
    cnorm[row] = s;
  }
}

// Async global->LDS copy of one 16B beat (no VGPR round-trip; tracked by ASYNCcnt).
__device__ __forceinline__ void async_b128(unsigned lds_byte_addr, const void* gaddr) {
  asm volatile("global_load_async_to_lds_b128 %0, %1, off"
               :: "v"(lds_byte_addr), "v"(gaddr)
               : "memory");
}
__device__ __forceinline__ void wait_asynccnt0() {
  asm volatile("s_wait_asynccnt 0x0" ::: "memory");
}

// ---------------- main: WMMA scores + running argmin ----------------
// 256 threads = 8 waves; each wave owns 16 rows of x; WG covers 128 rows.
__global__ __launch_bounds__(256) void vq_argmin(const float* __restrict__ x,
                                                 const uint4* __restrict__ chq,
                                                 const uint4* __restrict__ clq,
                                                 const float* __restrict__ cnorm,
                                                 int* __restrict__ idx_out) {
  __shared__ uint4 lds[2 * BUFQ];  // double buffer; each: [hi rows 0..15 | lo rows 0..15]

  const int tid  = threadIdx.x;
  const int lane = tid & 31;
  const int l16  = lane & 15;
  const int half = (lane >> 4) & 1;
  const int wave = tid >> 5;
  const long rowBase = (long)blockIdx.x * 128 + wave * 16;

  // Per-thread staging slice: row r16 (0..15), 32B segment seg (0..15); 4 b128 beats.
  const int r16 = tid >> 4;
  const int seg = tid & 15;
  const int dstHi = r16 * ROWQ + seg * 2;           // uint4 index within buffer
  const int dstLo = CHUNK * ROWQ + dstHi;

  // ---- Build resident A fragments (16 rows x 256 K, bf16 hi+lo) ----
  // A layout for 16x16x32 bf16: lane m(0..15) elements 0..15 = K {0..15} consecutive,
  // lanes 16..31 hold K {16..31} (matches B contiguous order).
  v16bf a_hi[8], a_lo[8];
  {
    const float* xr = x + (rowBase + l16) * DIMK;
#pragma unroll
    for (int kc = 0; kc < 8; ++kc) {
      const int k0 = kc * 32 + half * 16;
      const float4 f0 = *(const float4*)(xr + k0);
      const float4 f1 = *(const float4*)(xr + k0 + 4);
      const float4 f2 = *(const float4*)(xr + k0 + 8);
      const float4 f3 = *(const float4*)(xr + k0 + 12);
      float f[16];
      f[0]=f0.x; f[1]=f0.y; f[2]=f0.z;  f[3]=f0.w;
      f[4]=f1.x; f[5]=f1.y; f[6]=f1.z;  f[7]=f1.w;
      f[8]=f2.x; f[9]=f2.y; f[10]=f2.z; f[11]=f2.w;
      f[12]=f3.x; f[13]=f3.y; f[14]=f3.z; f[15]=f3.w;
#pragma unroll
      for (int i = 0; i < 16; ++i) {
        const __bf16 h = (__bf16)f[i];
        a_hi[kc][i] = h;
        a_lo[kc][i] = (__bf16)(f[i] - (float)h);
      }
    }
  }

  float best[8];
  int   bidx[8];
#pragma unroll
  for (int r = 0; r < 8; ++r) { best[r] = 3.4e38f; bidx[r] = 0; }

  // Prologue: stage chunk 0 into buffer 0 (async, no VGPR round-trip).
  {
    const int gsrc = r16 * 32 + seg * 2;   // codes row = 512B = 32 uint4
    async_b128((unsigned)(size_t)&lds[dstHi + 0], chq + gsrc + 0);
    async_b128((unsigned)(size_t)&lds[dstHi + 1], chq + gsrc + 1);
    async_b128((unsigned)(size_t)&lds[dstLo + 0], clq + gsrc + 0);
    async_b128((unsigned)(size_t)&lds[dstLo + 1], clq + gsrc + 1);
  }

  for (int cb = 0; cb < NCODES / CHUNK; ++cb) {
    const int bufOfs = (cb & 1) * BUFQ;

    wait_asynccnt0();   // my async beats for buf[cb&1] are in LDS
    __syncthreads();    // everyone's beats landed; everyone done reading buf[1-cur]

    if (cb + 1 < NCODES / CHUNK) {
      // Overlap next chunk's copy with this chunk's WMMAs.
      const int nOfs = ((cb + 1) & 1) * BUFQ;
      const int gsrc = ((cb + 1) * CHUNK + r16) * 32 + seg * 2;
      async_b128((unsigned)(size_t)&lds[nOfs + dstHi + 0], chq + gsrc + 0);
      async_b128((unsigned)(size_t)&lds[nOfs + dstHi + 1], chq + gsrc + 1);
      async_b128((unsigned)(size_t)&lds[nOfs + dstLo + 0], clq + gsrc + 0);
      async_b128((unsigned)(size_t)&lds[nOfs + dstLo + 1], clq + gsrc + 1);
    }

    const float cn = cnorm[cb * CHUNK + l16];  // L2-hot, 16 addrs/wave
    v8f acc = {};
#pragma unroll
    for (int kc = 0; kc < 8; ++kc) {
      // B layout (32x16 bf16): lane n(0..15) = K 0..15 consecutive, lanes 16..31 = K 16..31
      const int o = bufOfs + l16 * ROWQ + kc * 4 + half * 2;
      union { uint4 q[2]; v16bf v; } uh, ul;
      uh.q[0] = lds[o];                  uh.q[1] = lds[o + 1];
      ul.q[0] = lds[CHUNK * ROWQ + o];   ul.q[1] = lds[CHUNK * ROWQ + o + 1];
      // small terms first, then dominant term
      acc = __builtin_amdgcn_wmma_f32_16x16x32_bf16(false, a_lo[kc], false, uh.v,
                                                    (short)0, acc, false, false);
      acc = __builtin_amdgcn_wmma_f32_16x16x32_bf16(false, a_hi[kc], false, ul.v,
                                                    (short)0, acc, false, false);
      acc = __builtin_amdgcn_wmma_f32_16x16x32_bf16(false, a_hi[kc], false, uh.v,
                                                    (short)0, acc, false, false);
    }

    // C/D layout: VGPR r -> row (half*8 + r), lane%16 -> code column
    const int nglob = cb * CHUNK + l16;
#pragma unroll
    for (int r = 0; r < 8; ++r) {
      const float dist = cn - 2.0f * acc[r];
      if (dist < best[r]) { best[r] = dist; bidx[r] = nglob; }  // ascending cb => first-min kept
    }
  }

  // 16-lane butterfly min-reduce within each half (rows half*8 + r); tie -> lowest index
#pragma unroll
  for (int m = 1; m <= 8; m <<= 1) {
#pragma unroll
    for (int r = 0; r < 8; ++r) {
      const float ov = __shfl_xor(best[r], m, 32);
      const int   oi = __shfl_xor(bidx[r], m, 32);
      if (ov < best[r] || (ov == best[r] && oi < bidx[r])) { best[r] = ov; bidx[r] = oi; }
    }
  }

  if (l16 == 0) {
    const long ro = rowBase + half * 8;
#pragma unroll
    for (int r = 0; r < 8; ++r) idx_out[ro + r] = bidx[r];
  }
}

// ---------------- gather: quantized = codes[idx] ----------------
__global__ __launch_bounds__(256) void vq_gather(const float* __restrict__ codes,
                                                 const int* __restrict__ idx,
                                                 float* __restrict__ out) {
  const long row = (long)blockIdx.x * 4 + (threadIdx.x >> 6);  // 4 rows / block
  const int  c   = (threadIdx.x & 63) * 4;                     // 64 lanes x float4 = 256
  const int  id  = idx[row];
  const float4 v = *(const float4*)(codes + (long)id * DIMK + c);
  *(float4*)(out + row * DIMK + c) = v;
}

extern "C" void kernel_launch(void* const* d_in, const int* in_sizes, int n_in,
                              void* d_out, int out_size, void* d_ws, size_t ws_size,
                              hipStream_t stream) {
  const float* x     = (const float*)d_in[0];   // [8,8192,256] f32
  const float* codes = (const float*)d_in[1];   // [1024,256]   f32
  const long N = (long)in_sizes[0] / DIMK;      // 65536 rows

  float* outq = (float*)d_out;                  // quantized [N,256] f32
  int*   outi = (int*)(outq + N * (long)DIMK);  // indices   [N]     i32 (flat tail)

  __bf16* ch    = (__bf16*)d_ws;                        // 512 KB
  __bf16* cl    = ch + (size_t)NCODES * DIMK;           // 512 KB
  float*  cnorm = (float*)(cl + (size_t)NCODES * DIMK); // 4 KB
  (void)ws_size; (void)n_in; (void)out_size;

  vq_prep  <<<NCODES,         256, 0, stream>>>(codes, ch, cl, cnorm);
  vq_argmin<<<(int)(N / 128), 256, 0, stream>>>(x, (const uint4*)ch, (const uint4*)cl,
                                                cnorm, outi);
  vq_gather<<<(int)(N / 4),   256, 0, stream>>>(codes, outi, outq);
}
